// Mamba2DBlock_19104014532635
// MI455X (gfx1250) — matrix operand order repaired and checked by reading, not compiled, not run on hardware
//
#include <hip/hip_runtime.h>
#include <hip/hip_bf16.h>
#include <cstdint>
#include <cstddef>

// Problem constants (match reference)
#define BATCH  4
#define SEQ    1024
#define DMODEL 1024
#define DIN    2048          // D_INNER
#define DSTATE 16
#define DTR    64            // DT_RANK
#define MROWS  (BATCH*SEQ)   // 4096
#define XPROJ_N (DTR + 2*DSTATE)   // 96

typedef __attribute__((ext_vector_type(16))) __bf16 bf16x16;
typedef __attribute__((ext_vector_type(8)))  __bf16 bf16x8;
typedef __attribute__((ext_vector_type(8)))  float  f32x8;

__device__ __forceinline__ __bf16 f2bf(float f) {
  // round-to-nearest-even f32 -> bf16
  unsigned u = __builtin_bit_cast(unsigned, f);
  u += 0x7FFFu + ((u >> 16) & 1u);
  unsigned short s = (unsigned short)(u >> 16);
  return __builtin_bit_cast(__bf16, s);
}

// CDNA5 async global->LDS copy (ASYNCcnt-tracked). LDS byte address is the low
// 32 bits of the generic pointer (aperture rule, ISA 10.2).
__device__ __forceinline__ void async_copy_b128(const void* gsrc, void* lds_dst) {
  unsigned lds_addr = (unsigned)(uintptr_t)lds_dst;
  asm volatile("global_load_async_to_lds_b128 %0, %1, off"
               :: "v"(lds_addr), "v"(gsrc) : "memory");
}
__device__ __forceinline__ void wait_async0() {
  asm volatile("s_wait_asynccnt 0x0" ::: "memory");
}
__device__ __forceinline__ void wait_async2() {
  asm volatile("s_wait_asynccnt 0x2" ::: "memory");
}

// ---------------------------------------------------------------- converts
__global__ void k_f32_to_bf16(const float* __restrict__ in, __bf16* __restrict__ out, int n) {
  int i = blockIdx.x * 256 + threadIdx.x;
  if (i < n) out[i] = f2bf(in[i]);
}

// ---------------------------------------------------------------- layernorm
__global__ void k_layernorm(const float* __restrict__ x, const float* __restrict__ g,
                            const float* __restrict__ bta, __bf16* __restrict__ xn) {
  __shared__ float ssum[256], ssq[256];
  const int row = blockIdx.x;
  const float* xr = x + (size_t)row * DMODEL;
  float v[4], lsum = 0.f, lsq = 0.f;
#pragma unroll
  for (int i = 0; i < 4; i++) {
    v[i] = xr[threadIdx.x + 256 * i];
    lsum += v[i]; lsq += v[i] * v[i];
  }
  ssum[threadIdx.x] = lsum; ssq[threadIdx.x] = lsq;
  __syncthreads();
  for (int s = 128; s > 0; s >>= 1) {
    if (threadIdx.x < s) { ssum[threadIdx.x] += ssum[threadIdx.x + s]; ssq[threadIdx.x] += ssq[threadIdx.x + s]; }
    __syncthreads();
  }
  const float mu = ssum[0] * (1.0f / DMODEL);
  const float var = ssq[0] * (1.0f / DMODEL) - mu * mu;
  const float rs = rsqrtf(var + 1e-5f);
#pragma unroll
  for (int i = 0; i < 4; i++) {
    int c = threadIdx.x + 256 * i;
    xn[(size_t)row * DMODEL + c] = f2bf((v[i] - mu) * rs * g[c] + bta[c]);
  }
}

// ---------------------------------------------------------------- seq flip (per batch)
__global__ void k_flip(const __bf16* __restrict__ in, __bf16* __restrict__ out) {
  int i = blockIdx.x * 256 + threadIdx.x;           // over MROWS*DMODEL
  int c = i % DMODEL, row = i / DMODEL;
  int b = row / SEQ, l = row % SEQ;
  out[(size_t)(b * SEQ + (SEQ - 1 - l)) * DMODEL + c] = in[i];
}

// ================================================================
// LDS-staged bf16 WMMA GEMM (main path, N % 64 == 0, K % 64 == 0)
// C[m,n] = sum_k A[m,k]*W[n,k];  A:(MxK) bf16, W:(NxK) bf16, C:(MxN) f32
// Block: 256 threads (8 waves) -> 128(M) x 64(N) tile.
// Weight tile (64 rows x 64 K) double-buffered in LDS via async DMA.
// Per stage: load ALL fragments (2 A from global, 8 B from LDS) first,
// then issue the 8 WMMAs back-to-back (one wait per stage, not per op).
// EPI: 0 none, 1 softplus(v+bias[n]), 2 v+bias[n]+resid[m,n]
// ================================================================
#define BSTRIDE 72   // padded LDS row stride (elements): conflict-free, 16B-aligned

template <int EPI>
__global__ void __launch_bounds__(256)
k_gemm_lds(const __bf16* __restrict__ A, const __bf16* __restrict__ W,
           float* __restrict__ C, int M, int N, int K,
           const float* __restrict__ bias, const float* __restrict__ resid) {
  __shared__ __bf16 sB[2][64 * BSTRIDE];

  const int tid  = threadIdx.x;
  const int lane = tid & 31;
  const int wave = tid >> 5;
  const int m_tile = blockIdx.y * 128 + wave * 16;
  const int n_base = blockIdx.x * 64;
  const int m  = lane & 15;
  const int kh = lane >> 4;        // K-half this lane holds (A and B frags)
  const int nW = lane & 15;        // column this lane holds (B frag)

  const __bf16* arow = A + (size_t)(m_tile + m) * K;
  const int nblk = K >> 6;         // 64-wide K stages

  // stage issuer: 64 rows x 64 elems = 512 x 16B chunks, 2 per thread
  auto stage_B = [&](int buf, int kblk) {
#pragma unroll
    for (int i = 0; i < 2; i++) {
      int c = tid + i * 256;                 // 0..511
      int r = c >> 3, seg = c & 7;
      const __bf16* g = W + (size_t)(n_base + r) * K + (size_t)kblk * 64 + seg * 8;
      async_copy_b128(g, &sB[buf][r * BSTRIDE + seg * 8]);
    }
  };

  f32x8 acc[4];
#pragma unroll
  for (int t = 0; t < 4; t++)
#pragma unroll
    for (int r = 0; r < 8; r++) acc[t][r] = 0.0f;

  stage_B(0, 0);

  for (int blk = 0; blk < nblk; blk++) {
    const int cur = blk & 1;
    if (blk + 1 < nblk) { stage_B(1 - cur, blk + 1); wait_async2(); }
    else                { wait_async0(); }
    __syncthreads();

    // ---- load phase: all fragments of this 64-wide K stage ----
    bf16x16 a[2];
    bf16x16 b[2][4];
#pragma unroll
    for (int ks = 0; ks < 2; ks++) {
      const int kglob = blk * 64 + ks * 32;
      bf16x8 alo = *(const bf16x8*)(arow + kglob + kh * 16);
      bf16x8 ahi = *(const bf16x8*)(arow + kglob + kh * 16 + 8);
#pragma unroll
      for (int i = 0; i < 8; i++) { a[ks][i] = alo[i]; a[ks][i + 8] = ahi[i]; }
#pragma unroll
      for (int t = 0; t < 4; t++) {
        const __bf16* bp = &sB[cur][(t * 16 + nW) * BSTRIDE + ks * 32 + kh * 16];
        bf16x8 blo = *(const bf16x8*)(bp);
        bf16x8 bhi = *(const bf16x8*)(bp + 8);
#pragma unroll
        for (int i = 0; i < 8; i++) { b[ks][t][i] = blo[i]; b[ks][t][i + 8] = bhi[i]; }
      }
    }
    // ---- compute phase: 8 back-to-back WMMAs ----
#pragma unroll
    for (int ks = 0; ks < 2; ks++)
#pragma unroll
      for (int t = 0; t < 4; t++)
        acc[t] = __builtin_amdgcn_wmma_f32_16x16x32_bf16(
            false, a[ks], false, b[ks][t], (short)0, acc[t], false, false);

    __syncthreads();   // all waves done reading buf before it is re-filled
  }

#pragma unroll
  for (int t = 0; t < 4; t++) {
#pragma unroll
    for (int r = 0; r < 8; r++) {
      int row = m_tile + kh * 8 + r;
      int col = n_base + t * 16 + (lane & 15);
      float v = acc[t][r];
      if (EPI == 1) {
        v += bias[col];
        v = (v > 20.f) ? v : log1pf(__expf(v));      // softplus
      } else if (EPI == 2) {
        v += bias[col] + resid[(size_t)row * N + col];
      }
      C[(size_t)row * N + col] = v;
    }
  }
}

// ---------------------------------------------------------------- direct-global GEMM (small N path)
template <int NT, int EPI>
__global__ void __launch_bounds__(128)
k_gemm_bf16(const __bf16* __restrict__ A, const __bf16* __restrict__ W,
            float* __restrict__ C, int M, int N, int K,
            const float* __restrict__ bias, const float* __restrict__ resid) {
  const int lane = threadIdx.x & 31;
  const int wave = threadIdx.x >> 5;
  const int m_tile = (blockIdx.y * 4 + wave) * 16;
  const int n_base = blockIdx.x * (16 * NT);
  const int m  = lane & 15;
  const int kh = lane >> 4;
  const int nW = lane & 15;

  const __bf16* arow = A + (size_t)(m_tile + m) * K;

  f32x8 acc[NT];
#pragma unroll
  for (int t = 0; t < NT; t++)
#pragma unroll
    for (int r = 0; r < 8; r++) acc[t][r] = 0.0f;

  for (int k0 = 0; k0 < K; k0 += 32) {
    bf16x8 alo = *(const bf16x8*)(arow + k0 + kh * 16);
    bf16x8 ahi = *(const bf16x8*)(arow + k0 + kh * 16 + 8);
    bf16x16 a;
#pragma unroll
    for (int i = 0; i < 8; i++) { a[i] = alo[i]; a[i + 8] = ahi[i]; }

    const int kW = k0 + kh * 16;
#pragma unroll
    for (int t = 0; t < NT; t++) {
      bf16x16 b = *(const bf16x16*)(W + (size_t)(n_base + t * 16 + nW) * K + kW);
      acc[t] = __builtin_amdgcn_wmma_f32_16x16x32_bf16(
          false, a, false, b, (short)0, acc[t], false, false);
    }
  }

#pragma unroll
  for (int t = 0; t < NT; t++) {
#pragma unroll
    for (int r = 0; r < 8; r++) {
      int row = m_tile + kh * 8 + r;
      int col = n_base + t * 16 + (lane & 15);
      float v = acc[t][r];
      if (EPI == 1) {
        v += bias[col];
        v = (v > 20.f) ? v : log1pf(__expf(v));
      } else if (EPI == 2) {
        v += bias[col] + resid[(size_t)row * N + col];
      }
      C[(size_t)row * N + col] = v;
    }
  }
}

// ---------------------------------------------------------------- causal depthwise conv + SiLU
__global__ void k_conv_silu(const float* __restrict__ xz, const float* __restrict__ cw,
                            const float* __restrict__ cb, float* __restrict__ u,
                            __bf16* __restrict__ u_bf) {
  int i = blockIdx.x * 256 + threadIdx.x;           // MROWS*DIN
  int d = i % DIN, mrow = i / DIN;
  int l = mrow % SEQ, b = mrow / SEQ;
  float acc = cb[d];
#pragma unroll
  for (int j = 0; j < 4; j++) {
    int ls = l - 3 + j;
    if (ls >= 0) acc += xz[((size_t)(b * SEQ + ls)) * (2 * DIN) + d] * cw[d * 4 + j];
  }
  float s = acc / (1.f + __expf(-acc));             // silu
  u[i] = s;
  u_bf[i] = f2bf(s);
}

// ---------------------------------------------------------------- dt slice -> bf16
__global__ void k_slice_dt(const float* __restrict__ xdbl, __bf16* __restrict__ dt_bf) {
  int i = blockIdx.x * 256 + threadIdx.x;           // MROWS*DTR
  int r = i % DTR, mrow = i / DTR;
  dt_bf[i] = f2bf(xdbl[(size_t)mrow * XPROJ_N + r]);
}

// ---------------------------------------------------------------- selective scan (+ u*D + SiLU(z) gate)
__global__ void k_scan(const float* __restrict__ delta, const float* __restrict__ xdbl,
                       const float* __restrict__ u, const float* __restrict__ xz,
                       const float* __restrict__ A_log, const float* __restrict__ Dp,
                       __bf16* __restrict__ y_gate) {
  int idx = blockIdx.x * 256 + threadIdx.x;         // BATCH*DIN
  int d = idx % DIN, b = idx / DIN;
  float A[DSTATE], h[DSTATE];
#pragma unroll
  for (int n = 0; n < DSTATE; n++) { A[n] = -__expf(A_log[d * DSTATE + n]); h[n] = 0.f; }
  const float Dd = Dp[d];
  for (int l = 0; l < SEQ; l++) {
    const size_t row = (size_t)(b * SEQ + l);
    const float dt = delta[row * DIN + d];
    const float ut = u[row * DIN + d];
    const float* BC = xdbl + row * XPROJ_N + DTR;   // B[0..15], C[16..31]
    const float dtu = dt * ut;
    float y = 0.f;
#pragma unroll
    for (int n = 0; n < DSTATE; n++) {
      h[n] = __expf(dt * A[n]) * h[n] + dtu * BC[n];
      y += h[n] * BC[DSTATE + n];
    }
    const float zt = xz[row * (2 * DIN) + DIN + d];
    const float gate = zt / (1.f + __expf(-zt));
    y = (y + ut * Dd) * gate;
    y_gate[row * DIN + d] = f2bf(y);
  }
}

// ---------------------------------------------------------------- concat fwd + flipped bwd -> bf16
__global__ void k_concat(const float* __restrict__ yf, const float* __restrict__ yb,
                         __bf16* __restrict__ ycat) {
  int i = blockIdx.x * 256 + threadIdx.x;           // MROWS*2*DMODEL
  int c = i % (2 * DMODEL), mrow = i / (2 * DMODEL);
  int b = mrow / SEQ, l = mrow % SEQ;
  float v;
  if (c < DMODEL) v = yf[(size_t)mrow * DMODEL + c];
  else            v = yb[(size_t)(b * SEQ + (SEQ - 1 - l)) * DMODEL + (c - DMODEL)];
  ycat[i] = f2bf(v);
}

// ================================================================ host
extern "C" void kernel_launch(void* const* d_in, const int* in_sizes, int n_in,
                              void* d_out, int out_size, void* d_ws, size_t ws_size,
                              hipStream_t stream) {
  (void)in_sizes; (void)n_in; (void)out_size; (void)ws_size;
  const float* x       = (const float*)d_in[0];
  const float* norm_g  = (const float*)d_in[1];
  const float* norm_b  = (const float*)d_in[2];
  const float* merge_w = (const float*)d_in[3];
  const float* merge_b = (const float*)d_in[4];
  // per-direction inputs start at 5 (fwd) and 14 (bwd):
  // +0 in_w, +1 conv_w, +2 conv_b, +3 xproj_w, +4 dt_w, +5 dt_b, +6 A_log, +7 D, +8 out_w

  uint8_t* ws = (uint8_t*)d_ws;
  size_t off = 0;
  auto take = [&](size_t bytes) -> void* {
    void* p = ws + off;
    off = (off + bytes + 255) & ~(size_t)255;
    return p;
  };

  __bf16* xn  = (__bf16*)take((size_t)MROWS * DMODEL * 2);
  __bf16* xnf = (__bf16*)take((size_t)MROWS * DMODEL * 2);
  __bf16 *w_in[2], *w_xp[2], *w_dt[2], *w_out[2];
  for (int dir = 0; dir < 2; dir++) {
    w_in[dir]  = (__bf16*)take((size_t)(2 * DIN) * DMODEL * 2);
    w_xp[dir]  = (__bf16*)take((size_t)XPROJ_N * DIN * 2);
    w_dt[dir]  = (__bf16*)take((size_t)DIN * DTR * 2);
    w_out[dir] = (__bf16*)take((size_t)DMODEL * DIN * 2);
  }
  __bf16* w_mg  = (__bf16*)take((size_t)DMODEL * (2 * DMODEL) * 2);
  float*  xz    = (float*) take((size_t)MROWS * (2 * DIN) * 4);
  float*  u     = (float*) take((size_t)MROWS * DIN * 4);
  __bf16* u_bf  = (__bf16*)take((size_t)MROWS * DIN * 2);
  float*  xdbl  = (float*) take((size_t)MROWS * XPROJ_N * 4);
  __bf16* dtbf  = (__bf16*)take((size_t)MROWS * DTR * 2);
  float*  delta = (float*) take((size_t)MROWS * DIN * 4);
  __bf16* ygate = (__bf16*)take((size_t)MROWS * DIN * 2);
  float*  ydir0 = (float*) take((size_t)MROWS * DMODEL * 4);
  float*  ydir1 = (float*) take((size_t)MROWS * DMODEL * 4);
  __bf16* ycat  = (__bf16*)take((size_t)MROWS * (2 * DMODEL) * 2);

  auto cvt = [&](const float* src, __bf16* dst, size_t n) {
    k_f32_to_bf16<<<dim3((unsigned)((n + 255) / 256)), dim3(256), 0, stream>>>(src, dst, (int)n);
  };

  // weight down-converts (deterministic, graph-safe)
  cvt(merge_w, w_mg, (size_t)DMODEL * (2 * DMODEL));
  for (int dir = 0; dir < 2; dir++) {
    const int base = 5 + dir * 9;
    cvt((const float*)d_in[base + 0], w_in[dir],  (size_t)(2 * DIN) * DMODEL);
    cvt((const float*)d_in[base + 3], w_xp[dir],  (size_t)XPROJ_N * DIN);
    cvt((const float*)d_in[base + 4], w_dt[dir],  (size_t)DIN * DTR);
    cvt((const float*)d_in[base + 8], w_out[dir], (size_t)DMODEL * DIN);
  }

  k_layernorm<<<dim3(MROWS), dim3(256), 0, stream>>>(x, norm_g, norm_b, xn);
  k_flip<<<dim3((MROWS * DMODEL) / 256), dim3(256), 0, stream>>>(xn, xnf);

  for (int dir = 0; dir < 2; dir++) {
    const int base = 5 + dir * 9;
    const __bf16* xin   = dir ? xnf : xn;
    const float* conv_w = (const float*)d_in[base + 1];
    const float* conv_b = (const float*)d_in[base + 2];
    const float* dt_b   = (const float*)d_in[base + 5];
    const float* A_log  = (const float*)d_in[base + 6];
    const float* Dp     = (const float*)d_in[base + 7];
    float* ydir = dir ? ydir1 : ydir0;

    // in-proj: (4096 x 4096 x 1024), LDS-staged weights
    k_gemm_lds<0><<<dim3((2 * DIN) / 64, MROWS / 128), dim3(256), 0, stream>>>(
        xin, w_in[dir], xz, MROWS, 2 * DIN, DMODEL, nullptr, nullptr);
    // depthwise causal conv + silu
    k_conv_silu<<<dim3((MROWS * DIN) / 256), dim3(256), 0, stream>>>(xz, conv_w, conv_b, u, u_bf);
    // x-proj: (4096 x 96 x 2048), small-N direct path
    k_gemm_bf16<2, 0><<<dim3(XPROJ_N / 32, MROWS / 64), dim3(128), 0, stream>>>(
        u_bf, w_xp[dir], xdbl, MROWS, XPROJ_N, DIN, nullptr, nullptr);
    // dt slice -> bf16
    k_slice_dt<<<dim3((MROWS * DTR) / 256), dim3(256), 0, stream>>>(xdbl, dtbf);
    // delta = softplus(dt @ dt_w^T + dt_b): (4096 x 2048 x 64)
    k_gemm_lds<1><<<dim3(DIN / 64, MROWS / 128), dim3(256), 0, stream>>>(
        dtbf, w_dt[dir], delta, MROWS, DIN, DTR, dt_b, nullptr);
    // selective scan + u*D + silu(z) gate -> bf16
    k_scan<<<dim3((BATCH * DIN) / 256), dim3(256), 0, stream>>>(
        delta, xdbl, u, xz, A_log, Dp, ygate);
    // out-proj: (4096 x 1024 x 2048)
    k_gemm_lds<0><<<dim3(DMODEL / 64, MROWS / 128), dim3(256), 0, stream>>>(
        ygate, w_out[dir], ydir, MROWS, DMODEL, DIN, nullptr, nullptr);
  }

  // concat (unflip bwd) -> bf16
  k_concat<<<dim3((MROWS * 2 * DMODEL) / 256), dim3(256), 0, stream>>>(ydir0, ydir1, ycat);
  // merge: (4096 x 1024 x 2048) + bias + residual -> d_out (f32)
  k_gemm_lds<2><<<dim3(DMODEL / 64, MROWS / 128), dim3(256), 0, stream>>>(
      ycat, w_mg, (float*)d_out, MROWS, DMODEL, DIN, merge_b, x);
}
